// CustomRNN_49933289783733
// MI455X (gfx1250) — matrix operand order repaired and checked by reading, not compiled, not run on hardware
//
#include <hip/hip_runtime.h>

// Problem sizes (fixed by the reference)
#define B_   64
#define T_   256
#define V_   128
#define H_   1024
#define L_   2
#define NBLK 16
#define NTHR 256

typedef __attribute__((ext_vector_type(16))) __bf16 bf16x16;
typedef __attribute__((ext_vector_type(8)))  __bf16 bf16x8;
typedef __attribute__((ext_vector_type(8)))  float  f32x8;

union ABFrag { bf16x16 v; bf16x8 h2[2]; };

__device__ __forceinline__ float sig_(float x) { return 1.0f / (1.0f + __expf(-x)); }

// ---------------------------------------------------------------------------
// Grid-wide barrier: generation counter in global memory. sync[0] = arrival
// counter, sync[16] = generation (separate 64B line).
// ---------------------------------------------------------------------------
__device__ __forceinline__ void grid_barrier(unsigned* sync) {
  __syncthreads();
  if (threadIdx.x == 0) {
    __threadfence();
    volatile unsigned* gen = sync + 16;
    unsigned g = *gen;
    if (atomicAdd(sync, 1u) == NBLK - 1) {
      atomicExch(sync, 0u);
      __threadfence();
      atomicAdd((unsigned*)(sync + 16), 1u);
    } else {
      while (*gen == g) { __builtin_amdgcn_s_sleep(2); }
    }
    __threadfence();
  }
  __syncthreads();
}

// ---------------------------------------------------------------------------
// z[2 M-tiles, 4 gate tiles] += A(32xK) * W^T via bf16 WMMA.  K is a
// template constant so the 4 gate-row strides (g*H*K elements) fold into
// the 24-bit immediate offsets of a single load address stream.
// A frag per ISA: lane<16 holds K {k..k+7, k+16..k+23}; lane>=16 holds
// K {k+8..k+15, k+24..k+31}. B frag: lane = col; same K split as A
// (two 8-element runs along the weight row).
// Each B fragment feeds TWO wmma (both M tiles): halves weight traffic.
// All 12 loads issue before the 8 WMMAs so the scheduler can overlap with
// partial s_wait_loadcnt instead of a full drain per gate.
// ---------------------------------------------------------------------------
template <int K>
__device__ __forceinline__ void gemm_acc2(f32x8 acc[8],
                                          const __bf16* __restrict__ Abase,
                                          unsigned aoff0, unsigned aoff1,
                                          const __bf16* __restrict__ W,
                                          int col, int half) {
  const __bf16* wp = W + ((size_t)(unsigned)col * K + (unsigned)half * 8);
#pragma unroll 1
  for (int k = 0; k < K; k += 32) {
    ABFrag a0, a1;
    a0.h2[0] = *(const bf16x8*)(Abase + aoff0 + k);
    a0.h2[1] = *(const bf16x8*)(Abase + aoff0 + k + 16);
    a1.h2[0] = *(const bf16x8*)(Abase + aoff1 + k);
    a1.h2[1] = *(const bf16x8*)(Abase + aoff1 + k + 16);
    ABFrag b0_, b1_, b2_, b3_;
    b0_.h2[0] = *(const bf16x8*)(wp + k + (size_t)0 * H_ * K); b0_.h2[1] = *(const bf16x8*)(wp + k + 16 + (size_t)0 * H_ * K);
    b1_.h2[0] = *(const bf16x8*)(wp + k + (size_t)1 * H_ * K); b1_.h2[1] = *(const bf16x8*)(wp + k + 16 + (size_t)1 * H_ * K);
    b2_.h2[0] = *(const bf16x8*)(wp + k + (size_t)2 * H_ * K); b2_.h2[1] = *(const bf16x8*)(wp + k + 16 + (size_t)2 * H_ * K);
    b3_.h2[0] = *(const bf16x8*)(wp + k + (size_t)3 * H_ * K); b3_.h2[1] = *(const bf16x8*)(wp + k + 16 + (size_t)3 * H_ * K);
    bf16x16 bw0 = b0_.v, bw1 = b1_.v, bw2 = b2_.v, bw3 = b3_.v;
    acc[0] = __builtin_amdgcn_wmma_f32_16x16x32_bf16(false, a0.v, false, bw0, (short)0, acc[0], false, false);
    acc[4] = __builtin_amdgcn_wmma_f32_16x16x32_bf16(false, a1.v, false, bw0, (short)0, acc[4], false, false);
    acc[1] = __builtin_amdgcn_wmma_f32_16x16x32_bf16(false, a0.v, false, bw1, (short)0, acc[1], false, false);
    acc[5] = __builtin_amdgcn_wmma_f32_16x16x32_bf16(false, a1.v, false, bw1, (short)0, acc[5], false, false);
    acc[2] = __builtin_amdgcn_wmma_f32_16x16x32_bf16(false, a0.v, false, bw2, (short)0, acc[2], false, false);
    acc[6] = __builtin_amdgcn_wmma_f32_16x16x32_bf16(false, a1.v, false, bw2, (short)0, acc[6], false, false);
    acc[3] = __builtin_amdgcn_wmma_f32_16x16x32_bf16(false, a0.v, false, bw3, (short)0, acc[3], false, false);
    acc[7] = __builtin_amdgcn_wmma_f32_16x16x32_bf16(false, a1.v, false, bw3, (short)0, acc[7], false, false);
  }
}

// ---------------------------------------------------------------------------
// Gate combine + cell update for one 16x16 M tile. C/D layout: VGPR r,
// lanes 0-15 -> M=r, N=lane; lanes 16-31 -> M=8+r, N=lane-16.
// ---------------------------------------------------------------------------
__device__ __forceinline__ void cell_update(const f32x8* acc, float4 bias,
                                            float* __restrict__ cbuf,
                                            __bf16* __restrict__ hw,
                                            float* __restrict__ out,
                                            int t, int layer,
                                            int mbase, int half, int col) {
#pragma unroll
  for (int r = 0; r < 8; ++r) {
    const int b = mbase + half * 8 + r;
    const unsigned idx = (unsigned)b * H_ + col;
    float zi = acc[0][r] + bias.x;
    float zf = acc[1][r] + bias.y;
    float zg = acc[2][r] + bias.z;
    float zo = acc[3][r] + bias.w;
    float cn = sig_(zf) * cbuf[idx] + sig_(zi) * tanhf(zg);
    float hn = sig_(zo) * tanhf(cn);
    cbuf[idx] = cn;
    hw[idx] = (__bf16)hn;
    if (layer == 1) out[((size_t)b * T_ + t) * H_ + col] = hn;  // outs (B,T,H)
    if (t == T_ - 1) {
      const size_t HF = (size_t)B_ * T_ * H_;
      const size_t CF = HF + (size_t)L_ * B_ * H_;
      out[HF + (size_t)layer * B_ * H_ + idx] = hn;  // h_f (L,B,H)
      out[CF + (size_t)layer * B_ * H_ + idx] = cn;  // c_f (L,B,H)
    }
  }
}

// ---------------------------------------------------------------------------
// Persistent 2-layer LSTM. 16 blocks x 256 threads = 128 waves; each wave
// owns (mpair, j_tile): 32 batch rows x 16 hidden cols across all 4 gates
// (8 f32x8 accumulators = 64 VGPRs). Sequence handled in-kernel with grid
// barriers; h double-buffered, c updated in place.
// ---------------------------------------------------------------------------
__global__ __launch_bounds__(NTHR)
__attribute__((amdgpu_waves_per_eu(4)))
void lstm2_wmma(
    const __bf16* __restrict__ xb,    // (T,B,V) bf16
    const __bf16* __restrict__ Wih0,  // (4H,V) bf16
    const __bf16* __restrict__ Whh0,  // (4H,H) bf16
    const __bf16* __restrict__ Wih1,  // (4H,H) bf16
    const __bf16* __restrict__ Whh1,  // (4H,H) bf16
    const float* __restrict__ b0,     // (4H) combined bih0+bhh0
    const float* __restrict__ b1,     // (4H) combined bih1+bhh1
    __bf16* h0b, __bf16* h1b,         // 2 x (B,H) double buffers each
    float* c0, float* c1,             // (B,H) fp32 cell state
    float* out, unsigned* sync) {
  const int lane   = threadIdx.x & 31;
  const int wid    = (blockIdx.x * NTHR + threadIdx.x) >> 5;  // 0..127
  const int mp     = wid & 1;                                 // 2 M pairs
  const int jt     = wid >> 1;                                // 64 J tiles
  const int half   = lane >> 4;
  const int col    = jt * 16 + (lane & 15);                   // hidden col
  const int mbase0 = mp * 32;                                 // first M tile
  const int mbase1 = mp * 32 + 16;                            // second M tile
  const int row0   = mbase0 + (lane & 15);
  const int row1   = mbase1 + (lane & 15);

  // Per-lane 32-bit A offsets (element units) for K=V and K=H row strides.
  const unsigned av0 = (unsigned)row0 * V_ + half * 8;
  const unsigned av1 = (unsigned)row1 * V_ + half * 8;
  const unsigned ah0 = (unsigned)row0 * H_ + half * 8;
  const unsigned ah1 = (unsigned)row1 * H_ + half * 8;

  const float4 bias0 = {b0[col], b0[H_ + col], b0[2 * H_ + col], b0[3 * H_ + col]};
  const float4 bias1 = {b1[col], b1[H_ + col], b1[2 * H_ + col], b1[3 * H_ + col]};

  for (int t = 0; t < T_; ++t) {
    const int rb = t & 1, wb = rb ^ 1;
    const __bf16* h0r = h0b + (size_t)rb * B_ * H_;
    __bf16*       h0w = h0b + (size_t)wb * B_ * H_;
    const __bf16* h1r = h1b + (size_t)rb * B_ * H_;
    __bf16*       h1w = h1b + (size_t)wb * B_ * H_;

    {  // ---- layer 0: z = x_t @ Wih0^T + h0_{t-1} @ Whh0^T + b0 ----
      f32x8 acc[8] = {};
      gemm_acc2<V_>(acc, xb + (size_t)t * B_ * V_, av0, av1, Wih0, col, half);
      gemm_acc2<H_>(acc, h0r, ah0, ah1, Whh0, col, half);
      cell_update(acc + 0, bias0, c0, h0w, out, t, 0, mbase0, half, col);
      cell_update(acc + 4, bias0, c0, h0w, out, t, 0, mbase1, half, col);
    }
    grid_barrier(sync);

    {  // ---- layer 1: z = h0_t @ Wih1^T + h1_{t-1} @ Whh1^T + b1 ----
      f32x8 acc[8] = {};
      gemm_acc2<H_>(acc, h0w, ah0, ah1, Wih1, col, half);
      gemm_acc2<H_>(acc, h1r, ah0, ah1, Whh1, col, half);
      cell_update(acc + 0, bias1, c1, h1w, out, t, 1, mbase0, half, col);
      cell_update(acc + 4, bias1, c1, h1w, out, t, 1, mbase1, half, col);
    }
    grid_barrier(sync);
  }
}

// ---------------------------------------------------------------------------
// Prep kernels (one-time data movement: fp32 -> bf16, bias combine, init)
// ---------------------------------------------------------------------------
__global__ void k_cvt(const float* __restrict__ s, __bf16* __restrict__ d, int n) {
  for (int i = blockIdx.x * blockDim.x + threadIdx.x; i < n;
       i += gridDim.x * blockDim.x)
    d[i] = (__bf16)s[i];
}

__global__ void k_bias(const float* __restrict__ a, const float* __restrict__ b,
                       float* __restrict__ d, int n) {
  for (int i = blockIdx.x * blockDim.x + threadIdx.x; i < n;
       i += gridDim.x * blockDim.x)
    d[i] = a[i] + b[i];
}

__global__ void k_cvt_x(const float* __restrict__ x, __bf16* __restrict__ xb) {
  const int n = B_ * T_ * V_;
  for (int i = blockIdx.x * blockDim.x + threadIdx.x; i < n;
       i += gridDim.x * blockDim.x) {
    int v = i % V_;
    int t = (i / V_) % T_;
    int b = i / (V_ * T_);
    xb[((size_t)t * B_ + b) * V_ + v] = (__bf16)x[i];  // (B,T,V) -> (T,B,V)
  }
}

__global__ void k_init(const float* __restrict__ h, const float* __restrict__ c,
                       __bf16* h0b, __bf16* h1b, float* c0, float* c1,
                       unsigned* sync) {
  const int n = B_ * H_;
  int i0 = blockIdx.x * blockDim.x + threadIdx.x;
  if (i0 < 32) sync[i0] = 0u;
  for (int i = i0; i < n; i += gridDim.x * blockDim.x) {
    h0b[i] = (__bf16)h[i];        // layer 0 initial h -> buffer 0
    h1b[i] = (__bf16)h[n + i];    // layer 1 initial h -> buffer 0
    c0[i] = c[i];
    c1[i] = c[n + i];
  }
}

// ---------------------------------------------------------------------------
extern "C" void kernel_launch(void* const* d_in, const int* in_sizes, int n_in,
                              void* d_out, int out_size, void* d_ws,
                              size_t ws_size, hipStream_t stream) {
  (void)in_sizes; (void)n_in; (void)out_size; (void)ws_size;
  const float* x    = (const float*)d_in[0];
  const float* h    = (const float*)d_in[1];
  const float* c    = (const float*)d_in[2];
  const float* Wih0 = (const float*)d_in[3];
  const float* Whh0 = (const float*)d_in[4];
  const float* bih0 = (const float*)d_in[5];
  const float* bhh0 = (const float*)d_in[6];
  const float* Wih1 = (const float*)d_in[7];
  const float* Whh1 = (const float*)d_in[8];
  const float* bih1 = (const float*)d_in[9];
  const float* bhh1 = (const float*)d_in[10];
  float* out = (float*)d_out;

  char* ws = (char*)d_ws;
  size_t off = 0;
  auto take = [&](size_t bytes) -> char* {
    off = (off + 255) & ~(size_t)255;
    char* p = ws + off;
    off += bytes;
    return p;
  };

  unsigned* sync  = (unsigned*)take(256);
  float* b0       = (float*)take((size_t)4 * H_ * 4);
  float* b1       = (float*)take((size_t)4 * H_ * 4);
  __bf16* Wih0b   = (__bf16*)take((size_t)4 * H_ * V_ * 2);
  __bf16* Whh0b   = (__bf16*)take((size_t)4 * H_ * H_ * 2);
  __bf16* Wih1b   = (__bf16*)take((size_t)4 * H_ * H_ * 2);
  __bf16* Whh1b   = (__bf16*)take((size_t)4 * H_ * H_ * 2);
  __bf16* xb      = (__bf16*)take((size_t)T_ * B_ * V_ * 2);
  __bf16* h0b     = (__bf16*)take((size_t)2 * B_ * H_ * 2);
  __bf16* h1b     = (__bf16*)take((size_t)2 * B_ * H_ * 2);
  float* c0       = (float*)take((size_t)B_ * H_ * 4);
  float* c1       = (float*)take((size_t)B_ * H_ * 4);

  k_bias<<<8, 256, 0, stream>>>(bih0, bhh0, b0, 4 * H_);
  k_bias<<<8, 256, 0, stream>>>(bih1, bhh1, b1, 4 * H_);
  k_cvt<<<256, 256, 0, stream>>>(Wih0, Wih0b, 4 * H_ * V_);
  k_cvt<<<1024, 256, 0, stream>>>(Whh0, Whh0b, 4 * H_ * H_);
  k_cvt<<<1024, 256, 0, stream>>>(Wih1, Wih1b, 4 * H_ * H_);
  k_cvt<<<1024, 256, 0, stream>>>(Whh1, Whh1b, 4 * H_ * H_);
  k_cvt_x<<<512, 256, 0, stream>>>(x, xb);
  k_init<<<256, 256, 0, stream>>>(h, c, h0b, h1b, c0, c1, sync);

  lstm2_wmma<<<NBLK, NTHR, 0, stream>>>(xb, Wih0b, Whh0b, Wih1b, Whh1b, b0, b1,
                                        h0b, h1b, c0, c1, out, sync);
}
